// CrossAttentionLayer_65163243815121
// MI455X (gfx1250) — matrix operand order repaired and checked by reading, not compiled, not run on hardware
//
#include <hip/hip_runtime.h>
#include <hip/hip_bf16.h>
#include <math.h>

// ---------------- problem constants (from reference) ----------------
#define B_   8
#define C_   512
#define L_   256
#define HW_  4096
#define S_   4
#define P_   32              // pixels per workgroup tile
#define XST  528             // x-tile row stride in shorts (C_ + 16 pad)
#define LST  264             // k/q/v/ctx row stride in shorts (L_ + 8 pad)
#define EPS  1e-5f

typedef unsigned short u16;
typedef unsigned int   u32;
typedef __attribute__((ext_vector_type(16))) __bf16 v16bf;
typedef __attribute__((ext_vector_type(16))) u16    v16u;
typedef __attribute__((ext_vector_type(8)))  u16    v8u;
typedef __attribute__((ext_vector_type(4)))  u16    v4u;
typedef __attribute__((ext_vector_type(8)))  float  v8f;

// Pre-swizzled bf16 weights in d_ws: 16 matrices (4 streams x {Wk,Wq,Wd} + 4 x Wu),
// each 131072 elements, stored as WMMA A-fragments: frag f, lane l -> 16 contiguous
// bf16 at mat_base + f*512 + l*16  (one aligned 32B load per lane).
#define MAT_ELEMS (L_ * C_)          // 131072 (identical for Wu: 512*256)
#define WS_ELEMS  (16 * MAT_ELEMS)   // 2M bf16 = 4 MB

// LDS partition: x tile + k + q + v (bf16) + sim + BN coeffs (f32)
#define SM_X_SHORTS   (P_ * XST)            // 16896
#define SM_KQV_SHORTS (S_ * P_ * LST)       // 33792 each
#define SM_BYTES ((SM_X_SHORTS + 3 * SM_KQV_SHORTS) * 2 + (512 + 5 * 1024) * 4) // 259072

__device__ __forceinline__ u16 f2bf(float f) {
    u32 u = __builtin_bit_cast(u32, f);
    return (u16)((u + 0x7FFFu + ((u >> 16) & 1u)) >> 16);   // round-to-nearest-even
}
__device__ __forceinline__ float bf2f(u16 h) {
    u32 u = ((u32)h) << 16;
    return __builtin_bit_cast(float, u);
}

__device__ __forceinline__ v8f wmma_bf16(v16bf a, v16bf b, v8f c) {
    return __builtin_amdgcn_wmma_f32_16x16x32_bf16(false, a, false, b, (short)0, c, false, false);
}

// A fragment: one aligned 32B load from the pre-swizzled weight pool.
__device__ __forceinline__ v16bf load_a_pre(const u16* __restrict__ matbase, int frag, int lane) {
    return ((const v16bf*)matbase)[frag * 32 + lane];
}

// B fragment (32x16 bf16) from pixel-major LDS: lanes 0-15 -> K k0..k0+15 of
// column N=lane; lanes 16-31 -> K k0+16..k0+31 of column N=lane-16.
__device__ __forceinline__ v16bf load_b_frag(const u16* __restrict__ col, int k0, int lane) {
    const u16* src = col + k0 + ((lane & 16) ? 16 : 0);
    v16u bits;
#pragma unroll
    for (int i = 0; i < 16; ++i) bits[i] = src[i];          // -> 2x ds_load_b128
    return __builtin_bit_cast(v16bf, bits);
}

// ---------------- prologue: convert + swizzle weights to bf16 A-frag layout ----------------
// A layout (16x32 bf16): lane<16 holds row M=lane, K k0..k0+15;
// lane>=16 holds row M=lane-16, K k0+16..k0+31 (matches B fragment order).
__global__ __launch_bounds__(256)
void convert_weights_kernel(const float* __restrict__ Wk, const float* __restrict__ Wq,
                            const float* __restrict__ Wd, const float* __restrict__ Wu,
                            u16* __restrict__ ws) {
    const int gid    = blockIdx.x * 256 + threadIdx.x;  // 16 mats * 8192 lane-frags
    const int mat_id = gid >> 13;
    const int lf     = gid & 8191;
    const int frag   = lf >> 5;
    const int lane   = lf & 31;

    const float* src;
    int mtile, kk, rowstride;
    if (mat_id < 12) {                                  // s*3 + {Wk,Wq,Wd}
        const int s = mat_id / 3, m3 = mat_id % 3;
        src = (m3 == 0 ? Wk : (m3 == 1 ? Wq : Wd)) + (size_t)s * L_ * C_;
        rowstride = C_;
        mtile = frag >> 4;                              // 16 k-steps (C_/32)
        kk    = frag & 15;
    } else {                                            // Wu[s]
        const int s = mat_id - 12;
        src = Wu + (size_t)s * C_ * L_;
        rowstride = L_;
        mtile = frag >> 3;                              // 8 k-steps (L_/32)
        kk    = frag & 7;
    }
    const int rr = lane & 15;
    const float* row = src + (size_t)(mtile * 16 + rr) * rowstride + kk * 32 + ((lane & 16) ? 16 : 0);
    u16* dst = ws + (size_t)mat_id * MAT_ELEMS + (size_t)frag * 512 + lane * 16;

    v16u bits;
#pragma unroll
    for (int i = 0; i < 8; ++i) bits[i]     = f2bf(row[i]);
#pragma unroll
    for (int i = 0; i < 8; ++i) bits[8 + i] = f2bf(row[8 + i]);
    *(v16u*)dst = bits;                                 // 32B aligned store
}

// ---------------- fused cross-attention kernel ----------------
__global__ __launch_bounds__(256, 1)
void fused_xattn_kernel(const float* __restrict__ x0, const float* __restrict__ x1,
                        const float* __restrict__ x2, const float* __restrict__ x3,
                        const float* __restrict__ bk, const float* __restrict__ gk,
                        const float* __restrict__ betak, const float* __restrict__ mk,
                        const float* __restrict__ vk,
                        const float* __restrict__ bq, const float* __restrict__ gq,
                        const float* __restrict__ betaq, const float* __restrict__ mq,
                        const float* __restrict__ vq,
                        const float* __restrict__ bd, const float* __restrict__ bu,
                        const u16* __restrict__ wsw,
                        float* __restrict__ out) {
    extern __shared__ char smem_raw[];
    u16*   xbuf = (u16*)smem_raw;                 // [P_][XST]        bf16 x tile (one stream)
    u16*   kbuf = xbuf + SM_X_SHORTS;             // [S_][P_][LST]    k, later ctx
    u16*   qbuf = kbuf + SM_KQV_SHORTS;           // [S_][P_][LST]    q
    u16*   vbuf = qbuf + SM_KQV_SHORTS;           // [S_][P_][LST]    v
    float* simb = (float*)(vbuf + SM_KQV_SHORTS); // [P_][4][4]       sim -> attn
    float* cKa  = simb + 512;
    float* cKb  = cKa + S_ * L_;
    float* cQa  = cKb + S_ * L_;
    float* cQb  = cQa + S_ * L_;
    float* cD   = cQb + S_ * L_;

    const int tid  = threadIdx.x;
    const int lane = tid & 31;
    const int wid  = tid >> 5;
    const int b    = blockIdx.x >> 7;             // HW_/P_ = 128 tiles per image
    const int p0   = (blockIdx.x & 127) * P_;

    // ---- phase 0: fold BN into (a*y + b) coefficients in LDS ----
    for (int i = tid; i < S_ * L_; i += 256) {
        float sk = gk[i] * rsqrtf(vk[i] + EPS);
        cKa[i] = sk;
        cKb[i] = (bk[i] - mk[i]) * sk + betak[i];
        float sq = gq[i] * rsqrtf(vq[i] + EPS);
        cQa[i] = sq;
        cQb[i] = (bq[i] - mq[i]) * sq + betaq[i];
        cD[i]  = bd[i];
    }
    __syncthreads();

    // ---- phase 1: per stream, load x tile then k/q/v GEMMs on WMMA ----
    for (int s = 0; s < S_; ++s) {
        const float* xs = (s == 0 ? x0 : (s == 1 ? x1 : (s == 2 ? x2 : x3))) + (size_t)b * C_ * HW_ + p0;

        // transpose-load x tile [C_][P_] -> LDS [P_][C_] as bf16 (float4 reads)
        for (int idx = tid; idx < C_ * (P_ / 4); idx += 256) {
            int c = idx >> 3, p4 = (idx & 7) << 2;
            float4 xv = *(const float4*)(xs + (size_t)c * HW_ + p4);
            xbuf[(p4 + 0) * XST + c] = f2bf(xv.x);
            xbuf[(p4 + 1) * XST + c] = f2bf(xv.y);
            xbuf[(p4 + 2) * XST + c] = f2bf(xv.z);
            xbuf[(p4 + 3) * XST + c] = f2bf(xv.w);
        }
        __syncthreads();

        for (int mat = 0; mat < 3; ++mat) {
            const u16* matbase = wsw + (size_t)(s * 3 + mat) * MAT_ELEMS;
            u16*       dst     = (mat == 0 ? kbuf : (mat == 1 ? qbuf : vbuf)) + s * P_ * LST;

            for (int mi = 0; mi < 2; ++mi) {                 // 16 M-tiles over 8 waves
                const int m = wid * 2 + mi;
                __builtin_prefetch(matbase + (size_t)(m * 16) * 512, 0, 0); // global_prefetch_b8
                v8f acc0 = {0.f, 0.f, 0.f, 0.f, 0.f, 0.f, 0.f, 0.f};
                v8f acc1 = {0.f, 0.f, 0.f, 0.f, 0.f, 0.f, 0.f, 0.f};
#pragma unroll 8
                for (int kk = 0; kk < 16; ++kk) {            // K = C_ = 512
                    const int k0 = kk * 32;
                    v16bf af = load_a_pre(matbase, m * 16 + kk, lane);
                    v16bf b0 = load_b_frag(xbuf + (lane & 15) * XST, k0, lane);
                    v16bf b1 = load_b_frag(xbuf + ((lane & 15) + 16) * XST, k0, lane);
                    acc0 = wmma_bf16(af, b0, acc0);
                    acc1 = wmma_bf16(af, b1, acc1);
                }
                // epilogue: rows rb..rb+7 contiguous -> one 16B ds_store per pixel col
                const int rb = m * 16 + ((lane & 16) ? 8 : 0);
                const int pb = lane & 15;
                v8u pk0, pk1;
#pragma unroll
                for (int j = 0; j < 8; ++j) {
                    const int r = rb + j;
                    float v0 = acc0[j], v1 = acc1[j];
                    if (mat == 0) {
                        float a_ = cKa[s * L_ + r], b_ = cKb[s * L_ + r];
                        v0 = fmaxf(v0 * a_ + b_, 0.f);
                        v1 = fmaxf(v1 * a_ + b_, 0.f);
                    } else if (mat == 1) {
                        float a_ = cQa[s * L_ + r], b_ = cQb[s * L_ + r];
                        v0 = fmaxf(v0 * a_ + b_, 0.f);
                        v1 = fmaxf(v1 * a_ + b_, 0.f);
                    } else {
                        float b_ = cD[s * L_ + r];
                        v0 += b_;
                        v1 += b_;
                    }
                    pk0[j] = f2bf(v0);
                    pk1[j] = f2bf(v1);
                }
                *(v8u*)(dst + pb * LST + rb)        = pk0;   // ds_store_b128
                *(v8u*)(dst + (pb + 16) * LST + rb) = pk1;
            }
        }
        __syncthreads();   // k/q/v of stream s done; safe to reload xbuf
    }

    // ---- phase 2: sim[p][s][t] = (q[s]·k[t]) / sqrt(L) ----
    for (int task = tid; task < P_ * 16; task += 256) {
        const int p = task >> 4, st = task & 15, s = st >> 2, t = st & 3;
        const u16* qr = qbuf + (s * P_ + p) * LST;
        const u16* kr = kbuf + (t * P_ + p) * LST;
        float acc = 0.f;
        for (int l = 0; l < L_; l += 4) {
            v4u qa = *(const v4u*)(qr + l);
            v4u ka = *(const v4u*)(kr + l);
#pragma unroll
            for (int i = 0; i < 4; ++i) acc += bf2f(qa[i]) * bf2f(ka[i]);
        }
        simb[p * 16 + st] = acc * 0.0625f;    // L_^-0.5 = 1/16
    }
    __syncthreads();

    // ---- phase 3: softmax over t ----
    if (tid < P_ * 4) {
        float* row = simb + (tid >> 2) * 16 + (tid & 3) * 4;
        float mx = fmaxf(fmaxf(row[0], row[1]), fmaxf(row[2], row[3]));
        float e0 = __expf(row[0] - mx), e1 = __expf(row[1] - mx);
        float e2 = __expf(row[2] - mx), e3 = __expf(row[3] - mx);
        float inv = 1.f / (e0 + e1 + e2 + e3);
        row[0] = e0 * inv; row[1] = e1 * inv; row[2] = e2 * inv; row[3] = e3 * inv;
    }
    __syncthreads();

    // ---- phase 4: ctx[s] = sum_t attn[s][t] * v[t]; overwrite kbuf as ctx ----
    for (int idx = tid; idx < S_ * P_ * (L_ / 4); idx += 256) {
        const int l = (idx & 63) << 2;
        const int p = (idx >> 6) & 31;
        const int s = idx >> 11;
        const float* ar = simb + p * 16 + s * 4;
        const float a0 = ar[0], a1 = ar[1], a2 = ar[2], a3 = ar[3];
        v4u t0 = *(const v4u*)(vbuf + (0 * P_ + p) * LST + l);
        v4u t1 = *(const v4u*)(vbuf + (1 * P_ + p) * LST + l);
        v4u t2 = *(const v4u*)(vbuf + (2 * P_ + p) * LST + l);
        v4u t3 = *(const v4u*)(vbuf + (3 * P_ + p) * LST + l);
        v4u o;
#pragma unroll
        for (int i = 0; i < 4; ++i)
            o[i] = f2bf(a0 * bf2f(t0[i]) + a1 * bf2f(t1[i]) + a2 * bf2f(t2[i]) + a3 * bf2f(t3[i]));
        *(v4u*)(kbuf + (s * P_ + p) * LST + l) = o;
    }
    __syncthreads();

    // ---- phase 5: up-projection GEMM (C_ x L_) + residual, direct to out ----
    for (int s = 0; s < S_; ++s) {
        const u16*   matbase = wsw + (size_t)(12 + s) * MAT_ELEMS;
        const u16*   ctx     = kbuf + s * P_ * LST;
        const float* xs      = (s == 0 ? x0 : (s == 1 ? x1 : (s == 2 ? x2 : x3))) + (size_t)b * C_ * HW_ + p0;
        float*       op      = out + ((size_t)s * B_ + b) * C_ * HW_ + p0;

        for (int mi = 0; mi < 4; ++mi) {                     // 32 M-tiles over 8 waves
            const int m = wid * 4 + mi;
            __builtin_prefetch(matbase + (size_t)(m * 8) * 512, 0, 0);
            v8f acc0 = {0.f, 0.f, 0.f, 0.f, 0.f, 0.f, 0.f, 0.f};
            v8f acc1 = {0.f, 0.f, 0.f, 0.f, 0.f, 0.f, 0.f, 0.f};
#pragma unroll 8
            for (int kk = 0; kk < 8; ++kk) {                 // K = L_ = 256
                const int k0 = kk * 32;
                v16bf af = load_a_pre(matbase, m * 8 + kk, lane);
                v16bf b0 = load_b_frag(ctx + (lane & 15) * LST, k0, lane);
                v16bf b1 = load_b_frag(ctx + ((lane & 15) + 16) * LST, k0, lane);
                acc0 = wmma_bf16(af, b0, acc0);
                acc1 = wmma_bf16(af, b1, acc1);
            }
            const int rb = m * 16 + ((lane & 16) ? 8 : 0);
            const int pb = lane & 15;
#pragma unroll
            for (int j = 0; j < 8; ++j) {
                const int r = rb + j;
                const float bias = bu[s * C_ + r];
                op[r * HW_ + pb]      = acc0[j] + bias + xs[r * HW_ + pb];
                op[r * HW_ + pb + 16] = acc1[j] + bias + xs[r * HW_ + pb + 16];
            }
        }
    }
}

extern "C" void kernel_launch(void* const* d_in, const int* in_sizes, int n_in,
                              void* d_out, int out_size, void* d_ws, size_t ws_size,
                              hipStream_t stream) {
    const float* x0    = (const float*)d_in[0];
    const float* x1    = (const float*)d_in[1];
    const float* x2    = (const float*)d_in[2];
    const float* x3    = (const float*)d_in[3];
    const float* Wk    = (const float*)d_in[4];
    const float* bk    = (const float*)d_in[5];
    const float* gk    = (const float*)d_in[6];
    const float* betak = (const float*)d_in[7];
    const float* mk    = (const float*)d_in[8];
    const float* vk    = (const float*)d_in[9];
    const float* Wq    = (const float*)d_in[10];
    const float* bq    = (const float*)d_in[11];
    const float* gq    = (const float*)d_in[12];
    const float* betaq = (const float*)d_in[13];
    const float* mq    = (const float*)d_in[14];
    const float* vq    = (const float*)d_in[15];
    const float* Wd    = (const float*)d_in[16];
    const float* bd    = (const float*)d_in[17];
    const float* Wu    = (const float*)d_in[18];
    const float* bu    = (const float*)d_in[19];
    float* out = (float*)d_out;
    u16*   wsw = (u16*)d_ws;     // needs WS_ELEMS*2 = 4 MB of scratch

    // 1) convert + swizzle weights into WMMA A-fragment layout (bf16)
    hipLaunchKernelGGL(convert_weights_kernel, dim3(WS_ELEMS / 16 / 256), dim3(256), 0, stream,
                       Wk, Wq, Wd, Wu, wsw);

    // 2) fused cross-attention (stream-ordered after the converter)
    hipLaunchKernelGGL(fused_xattn_kernel, dim3(B_ * (HW_ / P_)), dim3(256), SM_BYTES, stream,
                       x0, x1, x2, x3, bk, gk, betak, mk, vk,
                       bq, gq, betaq, mq, vq, bd, bu, wsw, out);
}
